// Attention_72816875536915
// MI455X (gfx1250) — compile-verified, matrix operand order checked by reading, not run on hardware
//
#include <hip/hip_runtime.h>

typedef __attribute__((ext_vector_type(16))) __bf16 v16bf;
typedef __attribute__((ext_vector_type(8)))  __bf16 v8bf;
typedef __attribute__((ext_vector_type(8)))  float  v8f;
typedef __attribute__((ext_vector_type(4)))  unsigned int u32x4;
typedef __attribute__((ext_vector_type(8)))  int  i32x8;
typedef __attribute__((ext_vector_type(4)))  int  i32x4;

constexpr int Bc  = 2;
constexpr int Sc  = 2048;
constexpr int Dc  = 2048;
constexpr int Hc  = 16;
constexpr int HDc = 128;
constexpr int BSc = Bc * Sc;          // 4096

#if __has_builtin(__builtin_amdgcn_tensor_load_to_lds)
#define USE_TDM 1
#else
#define USE_TDM 0
#endif

// ---------------------------------------------------------------------------
// TDM: 2D tile load (bf16 elements) global -> LDS via Tensor Data Mover.
// Descriptor bit packing per CDNA5 ISA 8.3/8.4 (D# group0 128b, group1 256b).
// Issue from ONE wave only (EXEC is ignored by TENSOR_LOAD_TO_LDS).
// 6-arg builtin on this toolchain: (g0, g1, g2, g3, g4, cpol).
// ---------------------------------------------------------------------------
#if USE_TDM
__device__ __attribute__((always_inline)) inline void
tdm_load_2d(unsigned lds_off, const void* gaddr,
            unsigned tile_w, unsigned tile_h,
            unsigned tensor_w, unsigned tensor_h,
            unsigned row_stride_elems) {
    unsigned long long ga = (unsigned long long)(uintptr_t)gaddr;
    u32x4 g0;
    g0[0] = 1u;                                        // count=1 (valid user D#)
    g0[1] = lds_off;                                   // lds_addr (bytes)
    g0[2] = (unsigned)(ga & 0xffffffffu);              // global_addr[31:0]
    g0[3] = (unsigned)((ga >> 32) & 0x01ffffffu)       // global_addr[56:32]
            | (2u << 30);                              // type = 2 ("image")
    i32x8 g1;
    g1[0] = (int)(1u << 16);                           // data_size=1 -> 2 bytes
    g1[1] = (int)((tensor_w & 0xffffu) << 16);         // tensor_dim0[15:0] @48
    g1[2] = (int)((tensor_w >> 16) & 0xffffu)          // tensor_dim0[31:16]
            | (int)((tensor_h & 0xffffu) << 16);       // tensor_dim1[15:0]
    g1[3] = (int)((tensor_h >> 16) & 0xffffu)          // tensor_dim1[31:16]
            | (int)((tile_w & 0xffffu) << 16);         // tile_dim0 @112
    g1[4] = (int)(tile_h & 0xffffu);                   // tile_dim1 @128 (tile_dim2=0)
    g1[5] = (int)row_stride_elems;                     // tensor_dim0_stride[31:0]
    g1[6] = 0;
    g1[7] = 0;
    i32x4 z4 = {0, 0, 0, 0};
    i32x8 z8 = {0, 0, 0, 0, 0, 0, 0, 0};
    __builtin_amdgcn_tensor_load_to_lds(g0, g1, z4, z4, z8, 0);
}
#endif

__device__ __attribute__((always_inline)) inline unsigned lds_off_of(const void* p) {
    // LDS aperture: addr[63:32] selects the aperture, addr[31:0] is the offset.
    return (unsigned)(uintptr_t)p;
}

__device__ __attribute__((always_inline)) inline bool is_wave0() {
    return __builtin_amdgcn_readfirstlane((int)threadIdx.x) < 32;  // uniform branch
}

// ---------------------------------------------------------------------------
// Fragment loader for V_WMMA_*_16X16X32_BF16 (A and N-major B fragments).
// lane<16 -> row=lane, elems K{0..7} then K{16..23}; lane>=16 -> row=lane-16,
// elems K{8..15} then K{24..31}.
// ---------------------------------------------------------------------------
__device__ inline v16bf load_frag16(const __bf16* base, int ld) {
    int lane = threadIdx.x & 31;
    const __bf16* p = base + (size_t)(lane & 15) * ld + ((lane >> 4) << 3);
    v8bf lo = *(const v8bf*)p;
    v8bf hi = *(const v8bf*)(p + 16);
    return __builtin_shufflevector(lo, hi, 0,1,2,3,4,5,6,7,8,9,10,11,12,13,14,15);
}

__device__ inline v8f wmma_bf16(v16bf a, v16bf b, v8f c) {
    return __builtin_amdgcn_wmma_f32_16x16x32_bf16(false, a, false, b, (short)0, c,
                                                   false, false);
}

// ---------------------------------------------------------------------------
// fp32 -> bf16 cast
// ---------------------------------------------------------------------------
__global__ __launch_bounds__(256) void cast_bf16_kernel(const float* __restrict__ src,
                                                        __bf16* __restrict__ dst, int n) {
    int i = blockIdx.x * 256 + threadIdx.x;
    if (i < n) dst[i] = (__bf16)src[i];
}

// ---------------------------------------------------------------------------
// Cast + transpose a DxD fp32 weight into N-major bf16:  Wt[n][k] = W[k][n].
// ---------------------------------------------------------------------------
__global__ __launch_bounds__(256) void cast_transpose_kernel(const float* __restrict__ W,
                                                             __bf16* __restrict__ Wt) {
    __shared__ __bf16 tile[32][33];
    int n0 = blockIdx.x * 32, k0 = blockIdx.y * 32;
    int tx = threadIdx.x & 31, ty = threadIdx.x >> 5;
#pragma unroll
    for (int i = 0; i < 32; i += 8)
        tile[ty + i][tx] = (__bf16)W[(size_t)(k0 + ty + i) * Dc + n0 + tx];
    __syncthreads();
#pragma unroll
    for (int i = 0; i < 32; i += 8)
        Wt[(size_t)(n0 + ty + i) * Dc + k0 + tx] = tile[tx][ty + i];
}

// ---------------------------------------------------------------------------
// Per-(b,h) transpose of V: vtb[bh][d][s] = (bf16) vf[(b*S+s)][h*128+d].
// ---------------------------------------------------------------------------
__global__ __launch_bounds__(256) void transpose_v_kernel(const float* __restrict__ vf,
                                                          __bf16* __restrict__ vtb) {
    __shared__ __bf16 tile[32][33];
    int bh = blockIdx.z;
    int b = bh >> 4, h = bh & 15;
    int d0 = blockIdx.y * 32, s0 = blockIdx.x * 32;
    int tx = threadIdx.x & 31, ty = threadIdx.x >> 5;
#pragma unroll
    for (int i = 0; i < 32; i += 8)
        tile[ty + i][tx] =
            (__bf16)vf[(size_t)(b * Sc + s0 + ty + i) * Dc + h * HDc + d0 + tx];
    __syncthreads();
#pragma unroll
    for (int i = 0; i < 32; i += 8)
        vtb[((size_t)bh * HDc + d0 + ty + i) * Sc + s0 + tx] = tile[tx][ty + i];
}

// ---------------------------------------------------------------------------
// C(f32, 4096x2048) = A(bf16 row-major 4096x2048) * Bt(bf16 N-major 2048x2048)
// 256 threads / 8 waves, 128x128 block tile, K-step 32, TDM double buffering.
// ---------------------------------------------------------------------------
constexpr int GBM = 128, GBN = 128, GBK = 32;
constexpr int GM = BSc, GN = Dc, GK = Dc;

__global__ __launch_bounds__(256) void gemm_bf16_kernel(const __bf16* __restrict__ A,
                                                        const __bf16* __restrict__ Bt,
                                                        float* __restrict__ C) {
    __shared__ alignas(16) __bf16 As[2][GBM][GBK];   // 16 KB
    __shared__ alignas(16) __bf16 Bs[2][GBN][GBK];   // 16 KB (N-major)

    int tid = threadIdx.x;
    int w   = tid >> 5;
    int m0  = blockIdx.y * GBM;
    int n0  = blockIdx.x * GBN;
    int wm  = (w >> 1) * 32;
    int wn  = (w & 1) * 64;

    v8f acc[2][4] = {};

#if USE_TDM
    if (is_wave0()) {
        tdm_load_2d(lds_off_of(&As[0][0][0]), A + (size_t)m0 * GK, GBK, GBM, GK, GM, GK);
        tdm_load_2d(lds_off_of(&Bs[0][0][0]), Bt + (size_t)n0 * GK, GBK, GBN, GK, GN, GK);
    }
#endif

    int buf = 0;
    for (int k0 = 0; k0 < GK; k0 += GBK) {
        __syncthreads();   // all waves done with buffer buf^1
#if USE_TDM
        if (is_wave0()) {
            if (k0 + GBK < GK) {
                tdm_load_2d(lds_off_of(&As[buf ^ 1][0][0]),
                            A + (size_t)m0 * GK + (k0 + GBK), GBK, GBM, GK, GM, GK);
                tdm_load_2d(lds_off_of(&Bs[buf ^ 1][0][0]),
                            Bt + (size_t)n0 * GK + (k0 + GBK), GBK, GBN, GK, GN, GK);
                __builtin_amdgcn_s_wait_tensorcnt(2);  // in-order: current pair done
            } else {
                __builtin_amdgcn_s_wait_tensorcnt(0);
            }
        }
#else
        {   // fallback: contiguous-row staging (both tiles row/N-major)
            int r = tid >> 1, c = (tid & 1) * 16;
            const __bf16* ga = A + (size_t)(m0 + r) * GK + k0 + c;
            *(v8bf*)&As[buf][r][c]     = *(const v8bf*)ga;
            *(v8bf*)&As[buf][r][c + 8] = *(const v8bf*)(ga + 8);
            const __bf16* gb = Bt + (size_t)(n0 + r) * GK + k0 + c;
            *(v8bf*)&Bs[buf][r][c]     = *(const v8bf*)gb;
            *(v8bf*)&Bs[buf][r][c + 8] = *(const v8bf*)(gb + 8);
        }
#endif
        __syncthreads();   // publish buffer buf

        v16bf af[2], bf[4];
        af[0] = load_frag16(&As[buf][wm][0], GBK);
        af[1] = load_frag16(&As[buf][wm + 16][0], GBK);
#pragma unroll
        for (int j = 0; j < 4; ++j) bf[j] = load_frag16(&Bs[buf][wn + j * 16][0], GBK);
#pragma unroll
        for (int i = 0; i < 2; ++i)
#pragma unroll
            for (int j = 0; j < 4; ++j)
                acc[i][j] = wmma_bf16(af[i], bf[j], acc[i][j]);
        buf ^= 1;
    }

    // epilogue: one base address, all-constant offsets
    int lane = tid & 31;
    int col  = lane & 15;
    int rofs = (lane >> 4) * 8;
    float* cptr = C + (size_t)(m0 + wm + rofs) * GN + (n0 + wn + col);
#pragma unroll
    for (int i = 0; i < 2; ++i)
#pragma unroll
        for (int j = 0; j < 4; ++j)
#pragma unroll
            for (int r = 0; r < 8; ++r)
                cptr[(size_t)(i * 16 + r) * GN + j * 16] = acc[i][j][r];
}

// ---------------------------------------------------------------------------
// Fused per-head RMSNorm + RoPE for q and k. One block per (b*s, h), 128 thr.
// ---------------------------------------------------------------------------
__global__ __launch_bounds__(128) void rmsnorm_rope_kernel(const float* __restrict__ qf,
                                                           const float* __restrict__ kf,
                                                           const float* __restrict__ fc,
                                                           const float* __restrict__ fs,
                                                           const float* __restrict__ gq,
                                                           const float* __restrict__ gk,
                                                           __bf16* __restrict__ qb,
                                                           __bf16* __restrict__ kb) {
    int row = blockIdx.x;           // b*S + s
    int h   = blockIdx.y;
    int d   = threadIdx.x;
    size_t idx = (size_t)row * Dc + (size_t)h * HDc + d;

    float xq = qf[idx], xk = kf[idx];
    float sq = xq * xq, sk = xk * xk;
#pragma unroll
    for (int m = 16; m >= 1; m >>= 1) {
        sq += __shfl_xor(sq, m, 32);
        sk += __shfl_xor(sk, m, 32);
    }
    __shared__ float red[2][4];
    int w = d >> 5;
    if ((d & 31) == 0) { red[0][w] = sq; red[1][w] = sk; }
    __syncthreads();
    sq = red[0][0] + red[0][1] + red[0][2] + red[0][3];
    sk = red[1][0] + red[1][1] + red[1][2] + red[1][3];

    float rq = __frsqrt_rn(sq * (1.0f / HDc) + 1e-5f);
    float rk = __frsqrt_rn(sk * (1.0f / HDc) + 1e-5f);
    xq = xq * rq * gq[d];
    xk = xk * rk * gk[d];

    float c = fc[(size_t)row * (HDc / 2) + (d >> 1)];
    float s = fs[(size_t)row * (HDc / 2) + (d >> 1)];
    float pq = __shfl_xor(xq, 1, 32);
    float pk = __shfl_xor(xk, 1, 32);
    float oq, ok;
    if ((d & 1) == 0) { oq = xq * c - pq * s; ok = xk * c - pk * s; }
    else              { oq = pq * s + xq * c; ok = pk * s + xk * c; }

    qb[idx] = (__bf16)oq;
    kb[idx] = (__bf16)ok;
}

// ---------------------------------------------------------------------------
// Flash attention: grid (S/128, B*H), block 256 (8 waves). Each wave owns 16
// q rows; kv tiles of 64, double-buffered via TDM prefetch.
// ---------------------------------------------------------------------------
constexpr int QT = 128;
constexpr int KT = 64;

__global__ __launch_bounds__(256) void flash_attn_kernel(const __bf16* __restrict__ q,
                                                         const __bf16* __restrict__ k,
                                                         const __bf16* __restrict__ vtb,
                                                         float* __restrict__ ctx) {
    __shared__ alignas(16) __bf16 Qs[QT][HDc];         // 32 KB
    __shared__ alignas(16) __bf16 Ks[2][KT][HDc];      // 32 KB
    __shared__ alignas(16) __bf16 Vt[2][HDc][KT];      // 32 KB (d-major)
    __shared__ alignas(16) __bf16 Ps[8][16][KT];       // 16 KB per-wave P scratch

    int tid = threadIdx.x, w = tid >> 5, lane = tid & 31;
    int col = lane & 15, rofs = (lane >> 4) * 8;
    int q0 = blockIdx.x * QT;
    int bh = blockIdx.y;
    int b = bh >> 4, h = bh & 15;
    size_t baseOff = (size_t)b * Sc * Dc + (size_t)h * HDc;
    const __bf16* vhead = vtb + (size_t)bh * HDc * Sc;

    // stage Q tile (128x128) + first K/V tiles; wait for Q only (in-order)
#if USE_TDM
    if (is_wave0()) {
        tdm_load_2d(lds_off_of(&Qs[0][0]), q + baseOff + (size_t)q0 * Dc,
                    HDc, QT, Dc, Sc, Dc);
        tdm_load_2d(lds_off_of(&Ks[0][0][0]), k + baseOff, HDc, KT, Dc, Sc, Dc);
        tdm_load_2d(lds_off_of(&Vt[0][0][0]), vhead, KT, HDc, Sc, HDc, Sc);
        __builtin_amdgcn_s_wait_tensorcnt(2);   // Q complete
    }
#else
#pragma unroll
    for (int i = 0; i < 8; ++i) {
        int cidx = tid + i * 256;
        int r = cidx >> 4, c = (cidx & 15) << 3;
        *(v8bf*)&Qs[r][c] = *(const v8bf*)(q + baseOff + (size_t)(q0 + r) * Dc + c);
    }
#endif
    __syncthreads();

    v16bf af[4];
#pragma unroll
    for (int ks = 0; ks < 4; ++ks) af[ks] = load_frag16(&Qs[w * 16][ks * 32], HDc);

    float mrow[8], lrow[8];
    v8f   oacc[8] = {};
#pragma unroll
    for (int r = 0; r < 8; ++r) { mrow[r] = -3e38f; lrow[r] = 0.f; }

    const float scale = 0.08838834764831845f;   // 1/sqrt(128)

    int buf = 0;
    for (int kb0 = 0; kb0 < Sc; kb0 += KT) {
        __syncthreads();   // all waves done with buffer buf^1
#if USE_TDM
        if (is_wave0()) {
            if (kb0 + KT < Sc) {
                tdm_load_2d(lds_off_of(&Ks[buf ^ 1][0][0]),
                            k + baseOff + (size_t)(kb0 + KT) * Dc, HDc, KT, Dc, Sc, Dc);
                tdm_load_2d(lds_off_of(&Vt[buf ^ 1][0][0]),
                            vhead + (kb0 + KT), KT, HDc, Sc, HDc, Sc);
                __builtin_amdgcn_s_wait_tensorcnt(2);  // current K/V pair done
            } else {
                __builtin_amdgcn_s_wait_tensorcnt(0);
            }
        }
#else
#pragma unroll
        for (int i = 0; i < 4; ++i) {
            int cidx = tid + i * 256;
            int r = cidx >> 4, c = (cidx & 15) << 3;
            *(v8bf*)&Ks[buf][r][c] =
                *(const v8bf*)(k + baseOff + (size_t)(kb0 + r) * Dc + c);
            int rv = cidx >> 3, cv = (cidx & 7) << 3;
            *(v8bf*)&Vt[buf][rv][cv] =
                *(const v8bf*)(vhead + (size_t)rv * Sc + kb0 + cv);
        }
#endif
        __syncthreads();   // publish buffer buf

        // scores: 16x64 per wave, K = 128
        v8f sc[4];
#pragma unroll
        for (int ct = 0; ct < 4; ++ct) {
            v8f a = {};
#pragma unroll
            for (int ks = 0; ks < 4; ++ks) {
                v16bf bfrag = load_frag16(&Ks[buf][ct * 16][ks * 32], HDc);
                a = wmma_bf16(af[ks], bfrag, a);
            }
            sc[ct] = a;
        }

        // online softmax
        float alpha[8];
#pragma unroll
        for (int r = 0; r < 8; ++r) {
            float mx = -3e38f;
#pragma unroll
            for (int ct = 0; ct < 4; ++ct) {
                float val = sc[ct][r] * scale;
                sc[ct][r] = val;
                mx = fmaxf(mx, val);
            }
#pragma unroll
            for (int msk = 8; msk >= 1; msk >>= 1) mx = fmaxf(mx, __shfl_xor(mx, msk, 32));
            float mnew = fmaxf(mrow[r], mx);
            alpha[r] = __expf(mrow[r] - mnew);
            mrow[r]  = mnew;
            float rs = 0.f;
#pragma unroll
            for (int ct = 0; ct < 4; ++ct) {
                float p = __expf(sc[ct][r] - mnew);
                sc[ct][r] = p;
                rs += p;
            }
#pragma unroll
            for (int msk = 8; msk >= 1; msk >>= 1) rs += __shfl_xor(rs, msk, 32);
            lrow[r] = lrow[r] * alpha[r] + rs;
        }
#pragma unroll
        for (int dt = 0; dt < 8; ++dt)
#pragma unroll
            for (int r = 0; r < 8; ++r) oacc[dt][r] *= alpha[r];

        // P (C-layout) -> per-wave LDS scratch, row-major
#pragma unroll
        for (int ct = 0; ct < 4; ++ct)
#pragma unroll
            for (int r = 0; r < 8; ++r)
                Ps[w][rofs + r][ct * 16 + col] = (__bf16)sc[ct][r];
        __syncthreads();

        // PV: out 16x128 per wave, K = 64
#pragma unroll
        for (int kk = 0; kk < 2; ++kk) {
            v16bf pa = load_frag16(&Ps[w][0][kk * 32], KT);
#pragma unroll
            for (int dt = 0; dt < 8; ++dt) {
                v16bf vb = load_frag16(&Vt[buf][dt * 16][kk * 32], KT);
                oacc[dt] = wmma_bf16(pa, vb, oacc[dt]);
            }
        }
        buf ^= 1;
    }

    // epilogue: normalize, store fp32 context at constant offsets
    float* cp = ctx + baseOff + (size_t)(q0 + w * 16 + rofs) * Dc + col;
#pragma unroll
    for (int dt = 0; dt < 8; ++dt)
#pragma unroll
        for (int r = 0; r < 8; ++r)
            cp[(size_t)r * Dc + dt * 16] = oacc[dt][r] / lrow[r];
}

// ---------------------------------------------------------------------------
// Host-side launcher
// ---------------------------------------------------------------------------
extern "C" void kernel_launch(void* const* d_in, const int* in_sizes, int n_in,
                              void* d_out, int out_size, void* d_ws, size_t ws_size,
                              hipStream_t stream) {
    const float* X  = (const float*)d_in[0];
    const float* fc = (const float*)d_in[1];
    const float* fs = (const float*)d_in[2];
    const float* Wq = (const float*)d_in[3];
    const float* Wk = (const float*)d_in[4];
    const float* Wv = (const float*)d_in[5];
    const float* Wo = (const float*)d_in[6];
    const float* gq = (const float*)d_in[7];
    const float* gk = (const float*)d_in[8];
    float* out = (float*)d_out;

    char* ws = (char*)d_ws;
    size_t off = 0;
    auto carve = [&](size_t bytes) {
        void* p = ws + off;
        off += (bytes + 255) & ~(size_t)255;
        return p;
    };

    const size_t XE = (size_t)BSc * Dc;
    const size_t WE = (size_t)Dc * Dc;

    __bf16* Xb  = (__bf16*)carve(XE * 2);
    __bf16* Wqt = (__bf16*)carve(WE * 2);
    __bf16* Wkt = (__bf16*)carve(WE * 2);
    __bf16* Wvt = (__bf16*)carve(WE * 2);
    __bf16* Wot = (__bf16*)carve(WE * 2);
    float*  qf  = (float*)carve(XE * 4);
    float*  kf  = (float*)carve(XE * 4);
    float*  vf  = (float*)carve(XE * 4);
    __bf16* qb  = (__bf16*)carve(XE * 2);
    __bf16* kb  = (__bf16*)carve(XE * 2);
    __bf16* vtb = (__bf16*)carve(XE * 2);
    float*  ctx = (float*)carve(XE * 4);
    __bf16* ctb = (__bf16*)carve(XE * 2);

    // 1. cast X; cast+transpose weights to N-major
    cast_bf16_kernel<<<dim3((unsigned)((XE + 255) / 256)), dim3(256), 0, stream>>>(X, Xb, (int)XE);
    dim3 tgrid(Dc / 32, Dc / 32);
    cast_transpose_kernel<<<tgrid, dim3(256), 0, stream>>>(Wq, Wqt);
    cast_transpose_kernel<<<tgrid, dim3(256), 0, stream>>>(Wk, Wkt);
    cast_transpose_kernel<<<tgrid, dim3(256), 0, stream>>>(Wv, Wvt);
    cast_transpose_kernel<<<tgrid, dim3(256), 0, stream>>>(Wo, Wot);

    // 2. QKV projections
    dim3 ggrid(GN / GBN, GM / GBM);
    gemm_bf16_kernel<<<ggrid, dim3(256), 0, stream>>>(Xb, Wqt, qf);
    gemm_bf16_kernel<<<ggrid, dim3(256), 0, stream>>>(Xb, Wkt, kf);
    gemm_bf16_kernel<<<ggrid, dim3(256), 0, stream>>>(Xb, Wvt, vf);

    // 3. RMSNorm + RoPE (q,k -> bf16); transpose V per (b,h)
    rmsnorm_rope_kernel<<<dim3(BSc, Hc), dim3(128), 0, stream>>>(qf, kf, fc, fs, gq, gk, qb, kb);
    transpose_v_kernel<<<dim3(Sc / 32, HDc / 32, Bc * Hc), dim3(256), 0, stream>>>(vf, vtb);

    // 4. flash attention (double-buffered TDM kv pipeline)
    flash_attn_kernel<<<dim3(Sc / QT, Bc * Hc), dim3(256), 0, stream>>>(qb, kb, vtb, ctx);

    // 5. output projection
    cast_bf16_kernel<<<dim3((unsigned)((XE + 255) / 256)), dim3(256), 0, stream>>>(ctx, ctb, (int)XE);
    gemm_bf16_kernel<<<ggrid, dim3(256), 0, stream>>>(ctb, Wot, out);
}
